// MAB_64364379898215
// MI455X (gfx1250) — compile-verified, matrix operand order checked by reading, not run on hardware
//
#include <hip/hip_runtime.h>
#include <hip/hip_bf16.h>

// ---------------------------------------------------------------------------
// MAB transformer block for MI455X (gfx1250, wave32, WMMA f16->f32).
// B=4, nq=nk=1024, D=1024, H=16, HD=64.
// ---------------------------------------------------------------------------

typedef __attribute__((ext_vector_type(16))) _Float16 v16h;
typedef __attribute__((ext_vector_type(8)))  _Float16 v8h;
typedef __attribute__((ext_vector_type(4)))  _Float16 v4h;
typedef __attribute__((ext_vector_type(8)))  float    v8f;

#define FLAG_RELU  1
#define FLAG_TRANS 2

// CDNA5 16-bit A/B fragment (16 rows x 32 K), NT row-major source:
// lane l -> row (l&15); K groups {kb..kb+7} and {kb+16..kb+23}, kb=(l>>4)*8.
// Two 16-byte loads per fragment. Works for global or LDS pointers.
__device__ __forceinline__ v16h load_frag(const _Float16* base, int ld,
                                          int row0, int k0, int lane) {
  const int r  = row0 + (lane & 15);
  const int kb = k0 + ((lane >> 4) << 3);
  const _Float16* p = base + (size_t)r * ld + kb;
  v8h lo = *(const v8h*)(p);
  v8h hi = *(const v8h*)(p + 16);
  return __builtin_shufflevector(lo, hi, 0, 1, 2, 3, 4, 5, 6, 7,
                                         8, 9, 10, 11, 12, 13, 14, 15);
}

__device__ __forceinline__ v8f wmma_f16(v16h a, v16h b, v8f c) {
  return __builtin_amdgcn_wmma_f32_16x16x32_f16(false, a, false, b,
                                                (short)0, c, false, false);
}

// ---------------------------------------------------------------------------
// fp32 -> f16 conversion (vectorized x4)
// ---------------------------------------------------------------------------
__global__ __launch_bounds__(256) void cvt_f32_f16_kernel(
    const float* __restrict__ in, _Float16* __restrict__ out, int n4) {
  const int i = blockIdx.x * blockDim.x + threadIdx.x;
  if (i < n4) {
    float4 v = ((const float4*)in)[i];
    v4h o = {(_Float16)v.x, (_Float16)v.y, (_Float16)v.z, (_Float16)v.w};
    ((v4h*)out)[i] = o;
  }
}

// ---------------------------------------------------------------------------
// NT GEMM: C[m,n] = sum_k A[m,k] * Bw[n,k]  (A:[M,K], Bw:[N,K], row-major f16)
// Block = 4 waves (2x2) over a 128x128 C tile; each wave owns 64x64 (4x4 WMMA
// subtiles): 16 v_wmma per 32-k step vs 16 b128 fragment loads (1 load/WMMA).
// No LDS: fragments come straight from global; re-use is served by 192MB L2.
// Unconditional speculative prefetch keeps the loop branch-free.
// Epilogue: optional bias, ReLU, f32 store, f16 store (optionally transposed
// per-head for the V matrix: Vt[b][h][e][key]).
// ---------------------------------------------------------------------------
__global__ __launch_bounds__(128) void gemm_nt_kernel(
    const _Float16* __restrict__ A, const _Float16* __restrict__ Bw,
    float* __restrict__ Cf, _Float16* __restrict__ Ch,
    const float* __restrict__ bias, int Mdim, int Ndim, int Kdim, int flags) {
  const int lane = threadIdx.x & 31;
  const int wave = threadIdx.x >> 5;
  const int tiles_n = Ndim >> 7;
  const int bm = blockIdx.x / tiles_n;
  const int bn = blockIdx.x - bm * tiles_n;
  const int m0 = bm * 128 + (wave & 1) * 64;
  const int n0 = bn * 128 + (wave >> 1) * 64;

  v8f acc[4][4] = {};

#pragma unroll 1
  for (int k0 = 0; k0 < Kdim; k0 += 32) {
    // Speculative prefetch of next k-chunk (safe past-the-end; no fault).
    __builtin_prefetch(A + (size_t)(m0 + (lane & 15)) * Kdim + k0 + 32, 0, 1);
    __builtin_prefetch(Bw + (size_t)(n0 + (lane & 15)) * Kdim + k0 + 32, 0, 1);

    v16h af[4], bf[4];
#pragma unroll
    for (int i = 0; i < 4; i++) af[i] = load_frag(A, Kdim, m0 + 16 * i, k0, lane);
#pragma unroll
    for (int j = 0; j < 4; j++) bf[j] = load_frag(Bw, Kdim, n0 + 16 * j, k0, lane);
#pragma unroll
    for (int i = 0; i < 4; i++)
#pragma unroll
      for (int j = 0; j < 4; j++)
        acc[i][j] = wmma_f16(af[i], bf[j], acc[i][j]);
  }

  // C layout: lane (l&15) -> N; vgpr r -> M = r + 8*(l>>4)
  const int nl = lane & 15;
  const int mh = (lane >> 4) << 3;
#pragma unroll
  for (int i = 0; i < 4; i++) {
#pragma unroll
    for (int j = 0; j < 4; j++) {
      const int n = n0 + j * 16 + nl;
      const float bv = bias ? bias[n] : 0.f;
#pragma unroll
      for (int r = 0; r < 8; r++) {
        const int m = m0 + 16 * i + mh + r;
        float v = acc[i][j][r] + bv;
        if (flags & FLAG_RELU) v = fmaxf(v, 0.f);
        if (Cf) Cf[(size_t)m * Ndim + n] = v;
        if (Ch) {
          if (flags & FLAG_TRANS) {
            // Vt[b][h][e][key]: b=m>>10, key=m&1023, n=h*64+e
            Ch[((size_t)((m >> 10) << 10) + n) * 1024 + (m & 1023)] = (_Float16)v;
          } else {
            Ch[(size_t)m * Ndim + n] = (_Float16)v;
          }
        }
      }
    }
  }
}

// ---------------------------------------------------------------------------
// Flash attention: each wave owns 16 queries of one (b,h); streams 32-key
// blocks with online softmax. Per block: 4 WMMA for S=QK^T, LDS round-trip to
// convert P from C-layout to A-fragment layout, 4 WMMA for O += P V.
// Vt is pre-transposed ([b][h][e][key]) so PV fragments are row-major reads.
// ---------------------------------------------------------------------------
__global__ __launch_bounds__(256) void attn_kernel(
    const _Float16* __restrict__ Qh, const _Float16* __restrict__ Kh,
    const _Float16* __restrict__ Vt, const int* __restrict__ mask,
    _Float16* __restrict__ Oh) {
  __shared__ _Float16 plds[8][16 * 40];  // per-wave 16x32 P tile, ld=40 (pad)

  const int lane = threadIdx.x & 31;
  const int wave = threadIdx.x >> 5;
  const int bh   = blockIdx.x >> 3;  // 0..63
  const int qblk = blockIdx.x & 7;
  const int b = bh >> 4, h = bh & 15;
  const int q0 = qblk * 128 + wave * 16;

  const _Float16* Qbase = Qh + (size_t)b * 1024 * 1024 + h * 64;
  const _Float16* Kbase = Kh + (size_t)b * 1024 * 1024 + h * 64;
  const _Float16* Vbase = Vt + (size_t)(b * 16 + h) * 64 * 1024;
  const int* mb = mask + b * 1024;
  _Float16* myp = &plds[wave][0];

  const v16h qa0 = load_frag(Qbase, 1024, q0, 0, lane);
  const v16h qa1 = load_frag(Qbase, 1024, q0, 32, lane);

  v8f o[4] = {};
  v8f mrun = {-1e30f, -1e30f, -1e30f, -1e30f, -1e30f, -1e30f, -1e30f, -1e30f};
  v8f lrun = {};
  const float scale = 0.125f;  // 1/sqrt(64)
  const int mh8 = (lane >> 4) << 3;

#pragma unroll 1
  for (int kb = 0; kb < 1024; kb += 32) {
    v8f s[2] = {};
#pragma unroll
    for (int sub = 0; sub < 2; sub++) {
      v16h kf0 = load_frag(Kbase, 1024, kb + sub * 16, 0, lane);
      v16h kf1 = load_frag(Kbase, 1024, kb + sub * 16, 32, lane);
      s[sub] = wmma_f16(qa0, kf0, s[sub]);
      s[sub] = wmma_f16(qa1, kf1, s[sub]);
    }
    const int mk0 = mb[kb + (lane & 15)];
    const int mk1 = mb[kb + 16 + (lane & 15)];

    // per-row block max (rows live in vgpr elements; columns across 16 lanes)
    v8f bmax;
#pragma unroll
    for (int r = 0; r < 8; r++) {
      float v = -1e30f;
      if (mk0) v = fmaxf(v, s[0][r] * scale);
      if (mk1) v = fmaxf(v, s[1][r] * scale);
      bmax[r] = v;
    }
#pragma unroll
    for (int off = 1; off < 16; off <<= 1)
#pragma unroll
      for (int r = 0; r < 8; r++)
        bmax[r] = fmaxf(bmax[r], __shfl_xor(bmax[r], off, 32));

    v8f mnew, alpha, p0, p1, bsum;
#pragma unroll
    for (int r = 0; r < 8; r++) {
      mnew[r]  = fmaxf(mrun[r], bmax[r]);
      alpha[r] = __expf(mrun[r] - mnew[r]);
      p0[r] = mk0 ? __expf(s[0][r] * scale - mnew[r]) : 0.f;
      p1[r] = mk1 ? __expf(s[1][r] * scale - mnew[r]) : 0.f;
      bsum[r] = p0[r] + p1[r];
    }
#pragma unroll
    for (int off = 1; off < 16; off <<= 1)
#pragma unroll
      for (int r = 0; r < 8; r++)
        bsum[r] += __shfl_xor(bsum[r], off, 32);
#pragma unroll
    for (int r = 0; r < 8; r++) {
      lrun[r] = lrun[r] * alpha[r] + bsum[r];
      mrun[r] = mnew[r];
    }
#pragma unroll
    for (int t = 0; t < 4; t++)
#pragma unroll
      for (int r = 0; r < 8; r++) o[t][r] *= alpha[r];

    // C-layout -> A-fragment layout via private LDS tile (row q, col key)
#pragma unroll
    for (int r = 0; r < 8; r++) {
      myp[(r + mh8) * 40 + (lane & 15)]      = (_Float16)p0[r];
      myp[(r + mh8) * 40 + 16 + (lane & 15)] = (_Float16)p1[r];
    }
    asm volatile("s_wait_dscnt 0" ::: "memory");
    v16h pa = load_frag(myp, 40, 0, 0, lane);

#pragma unroll
    for (int t = 0; t < 4; t++) {
      v16h vf = load_frag(Vbase, 1024, t * 16, kb, lane);
      o[t] = wmma_f16(pa, vf, o[t]);
    }
  }

  // normalize; fully-masked rows (l==0) -> 0 (matches nan_to_num in ref)
#pragma unroll
  for (int t = 0; t < 4; t++) {
    const int e = t * 16 + (lane & 15);
#pragma unroll
    for (int r = 0; r < 8; r++) {
      const int q = q0 + mh8 + r;
      const float denom = lrun[r];
      const float val = denom > 0.f ? o[t][r] / denom : 0.f;
      Oh[((size_t)b * 1024 + q) * 1024 + h * 64 + e] = (_Float16)val;
    }
  }
}

// ---------------------------------------------------------------------------
// LayerNorm over D=1024: out = (a+b - mu)*rsqrt(var+eps)*g + beta.
// One 256-thread block per row; 4 elements/thread; shfl + LDS reduction.
// Emits f32 (+ optional f16 copy for the next f16 GEMM).
// ---------------------------------------------------------------------------
__global__ __launch_bounds__(256) void ln_kernel(
    const float* __restrict__ Ain, const float* __restrict__ Bin,
    const float* __restrict__ g, const float* __restrict__ beta,
    float* __restrict__ out32, _Float16* __restrict__ out16) {
  const int row = blockIdx.x;
  const int t = threadIdx.x;
  const int lane = t & 31, wave = t >> 5;

  const float4 xa = ((const float4*)(Ain + (size_t)row * 1024))[t];
  const float4 xb = ((const float4*)(Bin + (size_t)row * 1024))[t];
  float x[4] = {xa.x + xb.x, xa.y + xb.y, xa.z + xb.z, xa.w + xb.w};

  float s1 = x[0] + x[1] + x[2] + x[3];
  float s2 = x[0]*x[0] + x[1]*x[1] + x[2]*x[2] + x[3]*x[3];
#pragma unroll
  for (int off = 16; off >= 1; off >>= 1) {
    s1 += __shfl_xor(s1, off, 32);
    s2 += __shfl_xor(s2, off, 32);
  }
  __shared__ float r1[8], r2[8];
  if (lane == 0) { r1[wave] = s1; r2[wave] = s2; }
  __syncthreads();
  float tot1 = 0.f, tot2 = 0.f;
#pragma unroll
  for (int i = 0; i < 8; i++) { tot1 += r1[i]; tot2 += r2[i]; }

  const float mu   = tot1 * (1.f / 1024.f);
  const float var  = tot2 * (1.f / 1024.f) - mu * mu;
  const float rstd = rsqrtf(var + 1e-5f);
#pragma unroll
  for (int j = 0; j < 4; j++) {
    const int idx = t * 4 + j;
    const float y = (x[j] - mu) * rstd * g[idx] + beta[idx];
    out32[(size_t)row * 1024 + idx] = y;
    if (out16) out16[(size_t)row * 1024 + idx] = (_Float16)y;
  }
}

// ---------------------------------------------------------------------------
// Host-side orchestration
// ---------------------------------------------------------------------------
extern "C" void kernel_launch(void* const* d_in, const int* in_sizes, int n_in,
                              void* d_out, int out_size, void* d_ws, size_t ws_size,
                              hipStream_t stream) {
  (void)in_sizes; (void)n_in; (void)out_size; (void)ws_size;
  const float* X    = (const float*)d_in[0];
  const float* Y    = (const float*)d_in[1];
  const int*   mask = (const int*)d_in[2];
  const float* Wq   = (const float*)d_in[3];
  const float* Wk   = (const float*)d_in[4];
  const float* Wv   = (const float*)d_in[5];
  const float* Wo   = (const float*)d_in[6];
  const float* ln1g = (const float*)d_in[7];
  const float* ln1b = (const float*)d_in[8];
  const float* ln2g = (const float*)d_in[9];
  const float* ln2b = (const float*)d_in[10];
  const float* W1   = (const float*)d_in[11];
  const float* b1   = (const float*)d_in[12];
  const float* W2   = (const float*)d_in[13];
  const float* b2   = (const float*)d_in[14];

  char* ws = (char*)d_ws;
  const size_t MB = 1024 * 1024;
  // Workspace layout (84 MB) with lifetime-safe aliasing:
  _Float16* Xh   = (_Float16*)(ws + 0);        // 8MB; later reused as Oh
  _Float16* Yh   = (_Float16*)(ws + 8 * MB);   // 8MB; later reused as Hh
  _Float16* Wqh  = (_Float16*)(ws + 16 * MB);  // 2MB each
  _Float16* Wkh  = (_Float16*)(ws + 18 * MB);
  _Float16* Wvh  = (_Float16*)(ws + 20 * MB);
  _Float16* Woh  = (_Float16*)(ws + 22 * MB);
  _Float16* W1h  = (_Float16*)(ws + 24 * MB);
  _Float16* W2h  = (_Float16*)(ws + 26 * MB);
  _Float16* Qh   = (_Float16*)(ws + 28 * MB);  // 8MB; later reused as FF1h
  _Float16* Kh   = (_Float16*)(ws + 36 * MB);  // 8MB
  _Float16* Vt   = (_Float16*)(ws + 44 * MB);  // 8MB
  float*    Fbuf = (float*)(ws + 52 * MB);     // 16MB (Oproj, then FF2)
  float*    Hres = (float*)(ws + 68 * MB);     // 16MB
  _Float16* Ohb  = Xh;   // attn output (Xh dead after Q projection)
  _Float16* Hh   = Yh;   // ln1 f16 out (Yh dead after K/V projections)
  _Float16* FF1h = Qh;   // FFN1 out (Qh dead after attention)

  const int n4m = 4096 * 1024 / 4;  // X/Y element-quads
  const int n4w = 1024 * 1024 / 4;  // weight element-quads
  dim3 blk256(256);
  dim3 blk128(128);

  cvt_f32_f16_kernel<<<n4m / 256, blk256, 0, stream>>>(X, Xh, n4m);
  cvt_f32_f16_kernel<<<n4m / 256, blk256, 0, stream>>>(Y, Yh, n4m);
  cvt_f32_f16_kernel<<<n4w / 256, blk256, 0, stream>>>(Wq, Wqh, n4w);
  cvt_f32_f16_kernel<<<n4w / 256, blk256, 0, stream>>>(Wk, Wkh, n4w);
  cvt_f32_f16_kernel<<<n4w / 256, blk256, 0, stream>>>(Wv, Wvh, n4w);
  cvt_f32_f16_kernel<<<n4w / 256, blk256, 0, stream>>>(Wo, Woh, n4w);
  cvt_f32_f16_kernel<<<n4w / 256, blk256, 0, stream>>>(W1, W1h, n4w);
  cvt_f32_f16_kernel<<<n4w / 256, blk256, 0, stream>>>(W2, W2h, n4w);

  const int ggrid = (4096 / 128) * (1024 / 128);  // 256 workgroups

  // Q, K, V projections (V stored transposed per head)
  gemm_nt_kernel<<<ggrid, blk128, 0, stream>>>(Xh, Wqh, nullptr, Qh, nullptr,
                                               4096, 1024, 1024, 0);
  gemm_nt_kernel<<<ggrid, blk128, 0, stream>>>(Yh, Wkh, nullptr, Kh, nullptr,
                                               4096, 1024, 1024, 0);
  gemm_nt_kernel<<<ggrid, blk128, 0, stream>>>(Yh, Wvh, nullptr, Vt, nullptr,
                                               4096, 1024, 1024, FLAG_TRANS);

  // Flash attention: 64 (b,h) x 8 query-blocks
  attn_kernel<<<512, blk256, 0, stream>>>(Qh, Kh, Vt, mask, Ohb);

  // Output projection + LN1
  gemm_nt_kernel<<<ggrid, blk128, 0, stream>>>(Ohb, Woh, Fbuf, nullptr, nullptr,
                                               4096, 1024, 1024, 0);
  ln_kernel<<<4096, blk256, 0, stream>>>(X, Fbuf, ln1g, ln1b, Hres, Hh);

  // FFN + LN2
  gemm_nt_kernel<<<ggrid, blk128, 0, stream>>>(Hh, W1h, nullptr, FF1h, b1,
                                               4096, 1024, 1024, FLAG_RELU);
  gemm_nt_kernel<<<ggrid, blk128, 0, stream>>>(FF1h, W2h, Fbuf, nullptr, b2,
                                               4096, 1024, 1024, 0);
  ln_kernel<<<4096, blk256, 0, stream>>>(Hres, Fbuf, ln2g, ln2b, (float*)d_out,
                                         nullptr);
}
